// GptOssMoE_76656576299582
// MI455X (gfx1250) — compile-verified, matrix operand order checked by reading, not run on hardware
//
#include <hip/hip_runtime.h>
#include <hip/hip_bf16.h>
#include <math.h>

// ---------------- problem constants (match reference) ----------------
#define T_TOK 4096   // B*S
#define HDIM  2048
#define NEXP  32
#define TOPK  4
#define IDIM  1024

#define MT 64        // token tile (M): 4 WMMA subtiles
#define MSUB 4       // MT / 16
#define NT 128       // output-column tile (N)
#define KT 32        // WMMA K step (bf16 16x16x32)

#define LDSB_STRIDE 40     // shorts per LDS row (80B: 16B aligned, conflict-breaking)
#define ACT_STRIDE  1032   // shorts per activated row (2064B: 16B aligned)

typedef __attribute__((ext_vector_type(16))) __bf16 v16bf;
typedef __attribute__((ext_vector_type(4)))  __bf16 v4bf;
typedef __attribute__((ext_vector_type(8)))  float  v8f;

union V16BF {
    uint4 q[2];
    v16bf v;
};

// hardware bf16 convert (v_cvt_pk_bf16_f32, RNE) -- 4 floats -> 8B packed
__device__ __forceinline__ uint2 pack4(float a, float b, float c, float d) {
    union { v4bf v; uint2 q; } u;
    u.v[0] = (__bf16)a; u.v[1] = (__bf16)b; u.v[2] = (__bf16)c; u.v[3] = (__bf16)d;
    return u.q;
}
__device__ __forceinline__ unsigned short bf1(float a) {
    union { __bf16 b; unsigned short s; } u;
    u.b = (__bf16)a;
    return u.s;
}

// Store a 4(k) x 4(n) fp32 block transposed into LDS [n][k] bf16, 8B stores.
__device__ __forceinline__ void storeT4x4(unsigned short* lds, int n0, int k0,
                                          const float4 v[4]) {
    *(uint2*)&lds[(n0 + 0) * LDSB_STRIDE + k0] = pack4(v[0].x, v[1].x, v[2].x, v[3].x);
    *(uint2*)&lds[(n0 + 1) * LDSB_STRIDE + k0] = pack4(v[0].y, v[1].y, v[2].y, v[3].y);
    *(uint2*)&lds[(n0 + 2) * LDSB_STRIDE + k0] = pack4(v[0].z, v[1].z, v[2].z, v[3].z);
    *(uint2*)&lds[(n0 + 3) * LDSB_STRIDE + k0] = pack4(v[0].w, v[1].w, v[2].w, v[3].w);
}

__device__ __forceinline__ void load4rows(const float* p, int rowStride, float4 v[4]) {
    v[0] = *(const float4*)(p);
    v[1] = *(const float4*)(p + rowStride);
    v[2] = *(const float4*)(p + 2 * rowStride);
    v[3] = *(const float4*)(p + 3 * rowStride);
}

// ---------------- init / pre-pass kernels ----------------
__global__ void moe_zero_f32(float* p, int n) {
    int i = blockIdx.x * blockDim.x + threadIdx.x;
    if (i < n) p[i] = 0.0f;
}
__global__ void moe_zero_i32(int* p, int n) {
    int i = blockIdx.x * blockDim.x + threadIdx.x;
    if (i < n) p[i] = 0;
}
// x (fp32) -> xbf (bf16), 4 elements per thread
__global__ void moe_cvt_x(const float* __restrict__ x, unsigned short* __restrict__ xbf,
                          int n4) {
    int i = blockIdx.x * blockDim.x + threadIdx.x;
    if (i < n4) {
        float4 v = *(const float4*)(x + (size_t)i * 4);
        *(uint2*)(xbf + (size_t)i * 4) = pack4(v.x, v.y, v.z, v.w);
    }
}

// ---------------- gating: logits -> top4 -> softmax -> buckets ----------------
// one wave per token, lane = expert id (E == 32 == wave32)
__global__ void __launch_bounds__(256) moe_gate_kernel(
    const float* __restrict__ x, const float* __restrict__ gw,
    const float* __restrict__ gb, int* __restrict__ counts,
    int* __restrict__ btok, float* __restrict__ bwt)
{
    int wv   = threadIdx.x >> 5;
    int lane = threadIdx.x & 31;
    int t    = blockIdx.x * 8 + wv;
    if (t >= T_TOK) return;

    const float* xr = x  + (size_t)t * HDIM;
    const float* wr = gw + (size_t)lane * HDIM;
    float acc = 0.0f;
    for (int h = 0; h < HDIM; h += 4) {
        float4 xv  = *(const float4*)(xr + h);   // broadcast across lanes
        float4 wv4 = *(const float4*)(wr + h);
        acc += xv.x * wv4.x + xv.y * wv4.y + xv.z * wv4.z + xv.w * wv4.w;
    }
    acc += gb[lane];

    // iterative wave32 argmax for top-4 (ties -> lowest lane, matches stable top_k)
    float v = acc;
    float tw[TOPK]; int ti[TOPK];
    for (int k = 0; k < TOPK; k++) {
        float m = v; int mi = lane;
        for (int off = 16; off > 0; off >>= 1) {
            float ov = __shfl_xor(m, off, 32);
            int   oi = __shfl_xor(mi, off, 32);
            if (ov > m || (ov == m && oi < mi)) { m = ov; mi = oi; }
        }
        tw[k] = m; ti[k] = mi;
        if (lane == mi) v = -3.4e38f;
    }
    float mx = tw[0], s = 0.0f, p[TOPK];
    for (int k = 0; k < TOPK; k++) { p[k] = __expf(tw[k] - mx); s += p[k]; }
    float inv = 1.0f / s;

    if (lane < TOPK) {
        int e = ti[lane];
        int slot = atomicAdd(&counts[e], 1);
        btok[e * T_TOK + slot] = t;
        bwt [e * T_TOK + slot] = p[lane] * inv;
    }
}

// ---------------- fused expert kernel ----------------
// blockIdx.y = expert, blockIdx.x = M-tile (64 tokens) over the expert's bucket.
// 256 threads = 8 waves; wave w owns N-subtile n = w*16 for all 4 M-subtiles.
// A fragments load straight from the bf16 token buffer (b128s, L1/L2-resident);
// only B weight tiles go through LDS (transposed, bf16-converted on the fly).
__global__ void __launch_bounds__(256) moe_expert_kernel(
    const unsigned short* __restrict__ xbf,
    const float* __restrict__ w_gate, const float* __restrict__ b_gate,
    const float* __restrict__ w_up,   const float* __restrict__ b_up,
    const float* __restrict__ w_down, const float* __restrict__ b_down,
    const int* __restrict__ counts, const int* __restrict__ btok,
    const float* __restrict__ bwt, float* __restrict__ y)
{
    const int e     = blockIdx.y;
    const int mtile = blockIdx.x;
    const int cnt   = counts[e];
    if (mtile * MT >= cnt) return;

    const int tid  = threadIdx.x;
    const int lane = tid & 31;
    const int wv   = tid >> 5;        // 0..7
    const int lr   = lane & 15;       // row (A) / col (B) within subtile
    const int hv   = lane >> 4;       // K-half select
    const int nsub = wv * 16;         // wave's N offset within NT

    // dynamic LDS (~150KB; CDNA5 WGP has 320KB -> 2 workgroups/WGP)
    extern __shared__ __align__(16) unsigned char smem[];
    unsigned short* ldsB0  = (unsigned short*)smem;                  // [NT][LDSB_STRIDE]
    unsigned short* ldsB1  = ldsB0 + NT * LDSB_STRIDE;               // [NT][LDSB_STRIDE]
    unsigned short* ldsAct = ldsB1 + NT * LDSB_STRIDE;               // [MT][ACT_STRIDE]
    int*   tok_s = (int*)(ldsAct + MT * ACT_STRIDE);                 // [MT]
    float* rw_s  = (float*)(tok_s + MT);                             // [MT]

    if (tid < MT) {
        int r = mtile * MT + tid;
        if (r < cnt) { tok_s[tid] = btok[e * T_TOK + r]; rw_s[tid] = bwt[e * T_TOK + r]; }
        else         { tok_s[tid] = btok[e * T_TOK + mtile * MT]; rw_s[tid] = 0.0f; }
    }
    __syncthreads();

    // staging geometry (hoisted out of all loops)
    const int bn0 = (tid & 31) * 4;          // n offset of this thread's 4x4 B block
    const int bk0 = (tid >> 5) * 4;          // k offset of this thread's 4x4 B block
    const int fragB0 = (nsub + lr) * LDSB_STRIDE + hv * 8;

    // per-lane A rows (global bf16): lanes l and l+16 read token row 16*s + (l&15)
    const unsigned short* aRow[MSUB];
    #pragma unroll
    for (int s = 0; s < MSUB; s++)
        aRow[s] = xbf + (size_t)tok_s[16 * s + lr] * HDIM + hv * 8;

    // ---------------- phase 1: gate & up GEMMs + SwiGLU -> ldsAct ----------------
    for (int nb = 0; nb < IDIM; nb += NT) {
        v8f accg[MSUB], accu[MSUB];
        #pragma unroll
        for (int s = 0; s < MSUB; s++) {
            accg[s] = (v8f){0,0,0,0,0,0,0,0};
            accu[s] = (v8f){0,0,0,0,0,0,0,0};
        }

        // preload K-step 0 of B
        float4 g4[4], u4[4];
        {
            const float* pg = w_gate + (size_t)(e * HDIM + bk0) * IDIM + nb + bn0;
            const float* pu = w_up   + (size_t)(e * HDIM + bk0) * IDIM + nb + bn0;
            load4rows(pg, IDIM, g4);
            load4rows(pu, IDIM, u4);
        }

        for (int kb = 0; kb < HDIM; kb += KT) {
            // issue A fragment loads (global bf16, no LDS dependence)
            V16BF a[MSUB];
            #pragma unroll
            for (int s = 0; s < MSUB; s++) {
                a[s].q[0] = *(const uint4*)(aRow[s] + kb);
                a[s].q[1] = *(const uint4*)(aRow[s] + kb + 16);
            }

            // commit staged B tiles to LDS (hardware bf16 cvt)
            storeT4x4(ldsB0, bn0, bk0, g4);
            storeT4x4(ldsB1, bn0, bk0, u4);
            __syncthreads();

            // issue next K-step's B loads (overlaps WMMA below)
            int kn = kb + KT;
            if (kn < HDIM) {
                const float* pg = w_gate + (size_t)(e * HDIM + kn + bk0) * IDIM + nb + bn0;
                const float* pu = w_up   + (size_t)(e * HDIM + kn + bk0) * IDIM + nb + bn0;
                load4rows(pg, IDIM, g4);
                load4rows(pu, IDIM, u4);
            }

            V16BF bg, bu;
            bg.q[0] = *(const uint4*)&ldsB0[fragB0];
            bg.q[1] = *(const uint4*)&ldsB0[fragB0 + 16];
            bu.q[0] = *(const uint4*)&ldsB1[fragB0];
            bu.q[1] = *(const uint4*)&ldsB1[fragB0 + 16];

            #pragma unroll
            for (int s = 0; s < MSUB; s++) {
                accg[s] = __builtin_amdgcn_wmma_f32_16x16x32_bf16(
                    false, a[s].v, false, bg.v, (short)0, accg[s], false, false);
                accu[s] = __builtin_amdgcn_wmma_f32_16x16x32_bf16(
                    false, a[s].v, false, bu.v, (short)0, accu[s], false, false);
            }
            __syncthreads();
        }

        // SwiGLU (GPT-OSS: clamp 7, alpha 1.702) and park in ldsAct
        int col = nb + nsub + lr;
        float bgv = b_gate[e * IDIM + col];
        float buv = b_up  [e * IDIM + col];
        #pragma unroll
        for (int s = 0; s < MSUB; s++) {
            #pragma unroll
            for (int r = 0; r < 8; r++) {
                int m = 16 * s + r + hv * 8;
                float g = accg[s][r] + bgv;
                float u = accu[s][r] + buv;
                g = fminf(g, 7.0f);
                u = fminf(fmaxf(u, -7.0f), 7.0f);
                float glu = g * (1.0f / (1.0f + __expf(-1.702f * g)));
                ldsAct[m * ACT_STRIDE + col] = bf1((u + 1.0f) * glu);
            }
        }
    }
    __syncthreads();

    // ---------------- phase 2: down GEMM from ldsAct, weighted scatter-add ----------------
    for (int hb = 0; hb < HDIM; hb += NT) {
        v8f acc[MSUB];
        #pragma unroll
        for (int s = 0; s < MSUB; s++) acc[s] = (v8f){0,0,0,0,0,0,0,0};

        float4 d4[4];
        {
            const float* pd = w_down + (size_t)(e * IDIM + bk0) * HDIM + hb + bn0;
            load4rows(pd, HDIM, d4);
        }

        for (int kb = 0; kb < IDIM; kb += KT) {
            storeT4x4(ldsB0, bn0, bk0, d4);
            __syncthreads();

            int kn = kb + KT;
            if (kn < IDIM) {
                const float* pd = w_down + (size_t)(e * IDIM + kn + bk0) * HDIM + hb + bn0;
                load4rows(pd, HDIM, d4);
            }

            V16BF bd;
            bd.q[0] = *(const uint4*)&ldsB0[fragB0];
            bd.q[1] = *(const uint4*)&ldsB0[fragB0 + 16];

            #pragma unroll
            for (int s = 0; s < MSUB; s++) {
                V16BF a;
                int ao = (16 * s + lr) * ACT_STRIDE + kb + hv * 8;
                a.q[0] = *(const uint4*)&ldsAct[ao];
                a.q[1] = *(const uint4*)&ldsAct[ao + 16];
                acc[s] = __builtin_amdgcn_wmma_f32_16x16x32_bf16(
                    false, a.v, false, bd.v, (short)0, acc[s], false, false);
            }
            __syncthreads();
        }

        int col = hb + nsub + lr;
        float bdv = b_down[e * HDIM + col];
        #pragma unroll
        for (int s = 0; s < MSUB; s++) {
            #pragma unroll
            for (int r = 0; r < 8; r++) {
                int m = 16 * s + r + hv * 8;
                float o = (acc[s][r] + bdv) * rw_s[m];
                atomicAdd(&y[(size_t)tok_s[m] * HDIM + col], o);
            }
        }
    }
}

// ---------------- host launcher ----------------
extern "C" void kernel_launch(void* const* d_in, const int* in_sizes, int n_in,
                              void* d_out, int out_size, void* d_ws, size_t ws_size,
                              hipStream_t stream) {
    const float* x      = (const float*)d_in[0];
    const float* gw     = (const float*)d_in[1];
    const float* gb     = (const float*)d_in[2];
    const float* w_gate = (const float*)d_in[3];
    const float* b_gate = (const float*)d_in[4];
    const float* w_up   = (const float*)d_in[5];
    const float* b_up   = (const float*)d_in[6];
    const float* w_down = (const float*)d_in[7];
    const float* b_down = (const float*)d_in[8];
    float* y = (float*)d_out;

    // workspace: counts | btok | bwt | xbf(bf16)
    int*   counts = (int*)d_ws;                       // [NEXP]
    int*   btok   = counts + 32;                      // [NEXP * T_TOK]
    float* bwt    = (float*)(btok + NEXP * T_TOK);    // [NEXP * T_TOK]
    unsigned short* xbf = (unsigned short*)(bwt + NEXP * T_TOK);  // [T_TOK * HDIM]

    moe_zero_i32<<<1, 32, 0, stream>>>(counts, NEXP);
    {
        int n = T_TOK * HDIM;
        moe_zero_f32<<<(n + 255) / 256, 256, 0, stream>>>(y, n);
        int n4 = n / 4;
        moe_cvt_x<<<(n4 + 255) / 256, 256, 0, stream>>>(x, xbf, n4);
    }
    moe_gate_kernel<<<T_TOK / 8, 256, 0, stream>>>(x, gw, gb, counts, btok, bwt);

    // dynamic LDS: 2x B stage + 64-row activation tile + token/weight arrays
    size_t smem_bytes = (size_t)(2 * NT * LDSB_STRIDE + MT * ACT_STRIDE) *
                            sizeof(unsigned short) +
                        MT * (sizeof(int) + sizeof(float));
    dim3 grid(T_TOK / MT, NEXP);   // 64 M-tiles x 32 experts; empty tiles early-exit
    moe_expert_kernel<<<grid, 256, smem_bytes, stream>>>(xbf, w_gate, b_gate, w_up, b_up,
                                                         w_down, b_down, counts, btok,
                                                         bwt, y);
}
